// SparseRepPointsHead_11072425689886
// MI455X (gfx1250) — compile-verified, matrix operand order checked by reading, not run on hardware
//
#include <hip/hip_runtime.h>
#include <hip/hip_bf16.h>

// ---------------- problem constants ----------------
#define BB    8
#define DD    256
#define PP    9
#define KK    300
#define NCC   80
#define NO    900          // 3*K
#define PD    2304         // P*D
#define HCOLS 4608         // 2*P*D
#define BN_EPS 1e-5f

typedef __attribute__((ext_vector_type(16))) _Float16 v16h;
typedef __attribute__((ext_vector_type(8)))  float    v8f;

// ---------------- f32 -> f16 copy ----------------
__global__ void k_copy16(const float* __restrict__ s, _Float16* __restrict__ d, int n) {
    int t = blockIdx.x * blockDim.x + threadIdx.x;
    if (t < n) d[t] = (_Float16)s[t];
}

// transpose src(R,C) f32 -> dst(dstRows,R) f16, zero-pad rows >= C
__global__ void k_transpose16(const float* __restrict__ s, _Float16* __restrict__ d,
                              int R, int C, int dstRows) {
    int t = blockIdx.x * blockDim.x + threadIdx.x;
    if (t >= dstRows * R) return;
    int c = t / R, r = t % R;
    float v = (c < C) ? s[r * C + c] : 0.0f;
    d[(size_t)c * R + r] = (_Float16)v;
}

// ---------------- objectness head: conv(256->1,3x3)+BN+sigmoid ----------------
__global__ void k_obj_head(const float* __restrict__ x, const float* __restrict__ w,
                           const float* __restrict__ cb, const float* __restrict__ bng,
                           const float* __restrict__ bnb, const float* __restrict__ bnm,
                           const float* __restrict__ bnv, int lvl, int Hs, int Ws,
                           float* __restrict__ out) {
    int t = blockIdx.x * blockDim.x + threadIdx.x;
    int HW = Hs * Ws;
    if (t >= BB * HW) return;
    int b = t / HW, rr = t % HW, y = rr / Ws, xx = rr % Ws;
    float acc = cb[lvl];
    const float* wl = w + (size_t)lvl * DD * 9;
    for (int c = 0; c < DD; ++c) {
        const float* xp = x + ((size_t)(b * DD + c) * Hs) * Ws;
        const float* wp = wl + c * 9;
        #pragma unroll
        for (int dy = 0; dy < 3; ++dy) {
            int yy = y + dy - 1;
            if (yy < 0 || yy >= Hs) continue;
            #pragma unroll
            for (int dx = 0; dx < 3; ++dx) {
                int xc = xx + dx - 1;
                if (xc < 0 || xc >= Ws) continue;
                acc += xp[yy * Ws + xc] * wp[dy * 3 + dx];
            }
        }
    }
    float v = (acc - bnm[lvl]) * rsqrtf(bnv[lvl] + BN_EPS) * bng[lvl] + bnb[lvl];
    out[t] = 1.0f / (1.0f + expf(-v));
}

// ---------------- top-k (iterative argmax), one block per batch ----------------
__global__ __launch_bounds__(1024)
void k_topk(const float* __restrict__ p, int HW, int Hs, int Ws,
            int* __restrict__ idx_out, float* __restrict__ xy_out) {
    __shared__ float smax[1024];
    __shared__ int   sidx[1024];
    __shared__ int   swin;
    int b = blockIdx.x, tid = threadIdx.x;
    const float* pb = p + (size_t)b * HW;
    float v[16];
    #pragma unroll
    for (int i = 0; i < 16; ++i) {
        int pos = tid + i * 1024;
        v[i] = (pos < HW) ? pb[pos] : -1e30f;
    }
    for (int kk = 0; kk < KK; ++kk) {
        float m = -1e30f; int mi = 0;
        #pragma unroll
        for (int i = 0; i < 16; ++i)
            if (v[i] > m) { m = v[i]; mi = tid + i * 1024; }
        smax[tid] = m; sidx[tid] = mi;
        __syncthreads();
        for (int s = 512; s > 0; s >>= 1) {
            if (tid < s && smax[tid + s] > smax[tid]) {
                smax[tid] = smax[tid + s]; sidx[tid] = sidx[tid + s];
            }
            __syncthreads();
        }
        if (tid == 0) {
            int wi = sidx[0];
            swin = wi;
            idx_out[b * KK + kk] = wi;
            xy_out[(b * KK + kk) * 2 + 0] = (float)(wi / Ws) / (float)Hs;
            xy_out[(b * KK + kk) * 2 + 1] = (float)(wi % Ws) / (float)Ws;
        }
        __syncthreads();
        int wi = swin;
        #pragma unroll
        for (int i = 0; i < 16; ++i)
            if (tid + i * 1024 == wi) v[i] = -1e30f;
        __syncthreads();
    }
}

// ---------------- im2col patch gather at top-k positions, f16 ----------------
__global__ void k_gather_patches(const float* __restrict__ x, const int* __restrict__ idxs,
                                 int Hs, int Ws, _Float16* __restrict__ A) {
    int t = blockIdx.x * blockDim.x + threadIdx.x;
    if (t >= BB * KK * PD) return;
    int row = t / PD, col = t % PD;
    int b = row / KK, k = row % KK;
    int idx = idxs[b * KK + k];
    int c = col / 9, r9 = col % 9, dy = r9 / 3, dx = r9 % 3;
    int yy = idx / Ws + dy - 1, xc = idx % Ws + dx - 1;
    float v = 0.0f;
    if (yy >= 0 && yy < Hs && xc >= 0 && xc < Ws)
        v = x[((size_t)(b * DD + c) * Hs + yy) * Ws + xc];
    A[(size_t)row * PD + col] = (_Float16)v;
}

// ---------------- generic WMMA GEMM: C = act(A * Bt^T + bias) ----------------
// A:  (M,Kd)   f16 row-major, M   multiple of 16, Kd multiple of 32
// Bt: (N16,Kd) f16 row-major (B transposed), N16 multiple of 64 (zero-padded)
// One wave owns a 16x64 tile (4 N-tiles, A reused); branch-free inner loop,
// wave-uniform tile coords in SGPRs, f32 accumulation.
__global__ __launch_bounds__(256)
void k_wmma_gemm(const _Float16* __restrict__ A, const _Float16* __restrict__ Bt,
                 const float* __restrict__ bias, _Float16* __restrict__ outH,
                 float* __restrict__ outF, int M, int N16, int Kd,
                 int lda, int ldb, int ldo, int Nreal, int act) {
    int wave = threadIdx.x >> 5;
    int lane = threadIdx.x & 31;
    int tM = M >> 4, tN64 = N16 >> 6;
    int tile = __builtin_amdgcn_readfirstlane(blockIdx.x * 8 + wave);
    if (tile >= tM * tN64) return;
    int tm  = tile % tM;
    int tn0 = (tile / tM) * 4;
    int l16 = lane & 15, g = lane >> 4;

    const _Float16* arow = A  + (size_t)(tm * 16 + l16) * lda + g * 8;
    const _Float16* br0  = Bt + (size_t)(tn0 * 16 + l16) * ldb + g * 8;
    const _Float16* br1  = br0 + (size_t)16 * ldb;
    const _Float16* br2  = br1 + (size_t)16 * ldb;
    const _Float16* br3  = br2 + (size_t)16 * ldb;

    v8f acc0 = {}, acc1 = {}, acc2 = {}, acc3 = {};
    for (int kk = 0; kk < Kd; kk += 32) {
        v16h a, b0, b1, b2, b3;
        ((float4*)&a)[0]  = *(const float4*)(arow + kk);
        ((float4*)&a)[1]  = *(const float4*)(arow + kk + 16);
        ((float4*)&b0)[0] = *(const float4*)(br0 + kk);
        ((float4*)&b0)[1] = *(const float4*)(br0 + kk + 16);
        ((float4*)&b1)[0] = *(const float4*)(br1 + kk);
        ((float4*)&b1)[1] = *(const float4*)(br1 + kk + 16);
        ((float4*)&b2)[0] = *(const float4*)(br2 + kk);
        ((float4*)&b2)[1] = *(const float4*)(br2 + kk + 16);
        ((float4*)&b3)[0] = *(const float4*)(br3 + kk);
        ((float4*)&b3)[1] = *(const float4*)(br3 + kk + 16);
        if (kk + 64 < Kd) __builtin_prefetch(arow + kk + 64, 0, 1);
        acc0 = __builtin_amdgcn_wmma_f32_16x16x32_f16(false, a, false, b0, (short)0, acc0, false, false);
        acc1 = __builtin_amdgcn_wmma_f32_16x16x32_f16(false, a, false, b1, (short)0, acc1, false, false);
        acc2 = __builtin_amdgcn_wmma_f32_16x16x32_f16(false, a, false, b2, (short)0, acc2, false, false);
        acc3 = __builtin_amdgcn_wmma_f32_16x16x32_f16(false, a, false, b3, (short)0, acc3, false, false);
    }

    #pragma unroll
    for (int j = 0; j < 4; ++j) {
        const v8f& acc = (j == 0) ? acc0 : (j == 1) ? acc1 : (j == 2) ? acc2 : acc3;
        int col = (tn0 + j) * 16 + l16;
        if (col >= Nreal) continue;
        float bv = bias ? bias[col] : 0.0f;
        #pragma unroll
        for (int r = 0; r < 8; ++r) {
            int row = tm * 16 + r + 8 * g;
            float v = acc[r] + bv;
            if (act == 1) v = fmaxf(v, 0.0f);
            else if (act == 2) v = 1.0f / (1.0f + expf(-v));
            if (outH) outH[(size_t)row * ldo + col] = (_Float16)v;
            if (outF) outF[(size_t)row * ldo + col] = v;
        }
    }
}

// ---------------- 18-ch projection + tanh grid ----------------
__global__ void k_offsets_to_grid(const _Float16* __restrict__ offc,
                                  const float* __restrict__ ow, const float* __restrict__ ob,
                                  const float* __restrict__ xys, float* __restrict__ grid) {
    int t = blockIdx.x * blockDim.x + threadIdx.x;
    if (t >= BB * KK * PP) return;
    int row = t / PP, pp = t % PP;
    float d0 = ob[2 * pp], d1 = ob[2 * pp + 1];
    const _Float16* f = offc + (size_t)row * DD;
    const float* w0 = ow + (2 * pp) * DD;
    const float* w1 = w0 + DD;
    for (int c = 0; c < DD; ++c) {
        float fv = (float)f[c];
        d0 += w0[c] * fv;
        d1 += w1[c] * fv;
    }
    float xy0 = xys[row * 2 + 0], xy1 = xys[row * 2 + 1];
    grid[((size_t)row * PP + pp) * 2 + 0] = tanhf(d0 + xy0);
    grid[((size_t)row * PP + pp) * 2 + 1] = tanhf(d1 + xy1);
}

// ---------------- bilinear grid_sample -> h (f16, point-feature half) ----------------
__global__ void k_sample_to_h(const float* __restrict__ x, const float* __restrict__ grid,
                              int lvl, int Hs, int Ws, _Float16* __restrict__ h) {
    int t = blockIdx.x * blockDim.x + threadIdx.x;
    if (t >= BB * KK * PP * DD) return;
    int d = t & 255;
    int rest = t >> 8;
    int pp = rest % PP; rest /= PP;
    int k = rest % KK;
    int b = rest / KK;
    const float* gp = grid + ((size_t)(b * KK + k) * PP + pp) * 2;
    float ix = (gp[0] + 1.0f) * (Ws * 0.5f) - 0.5f;
    float iy = (gp[1] + 1.0f) * (Hs * 0.5f) - 0.5f;
    float x0f = floorf(ix), y0f = floorf(iy);
    float wx1 = ix - x0f, wx0 = 1.0f - wx1;
    float wy1 = iy - y0f, wy0 = 1.0f - wy1;
    int x0 = (int)x0f, y0 = (int)y0f;
    const float* plane = x + ((size_t)(b * DD + d) * Hs) * Ws;
    float acc = 0.0f;
    #pragma unroll
    for (int cy = 0; cy < 2; ++cy) {
        int yq = y0 + cy;
        if (yq < 0 || yq >= Hs) continue;
        float wy = cy ? wy1 : wy0;
        #pragma unroll
        for (int cx = 0; cx < 2; ++cx) {
            int xq = x0 + cx;
            if (xq < 0 || xq >= Ws) continue;
            float wx = cx ? wx1 : wx0;
            acc += plane[yq * Ws + xq] * wx * wy;
        }
    }
    h[(size_t)(b * NO + lvl * KK + k) * HCOLS + pp * DD + d] = (_Float16)acc;
}

// ---------------- sine positional embedding -> h (f16, pos half) ----------------
__global__ void k_sine_to_h(_Float16* __restrict__ h) {
    int t = blockIdx.x * blockDim.x + threadIdx.x;
    if (t >= KK * PP * DD) return;
    int d = t & 255;
    int pp = (t >> 8) % PP;
    int k = t / (DD * PP);
    const float scale = 6.28318530717958647692f;
    float base; int j;
    if (d < 128) { base = (float)(k + 1) / (300.0f + 1e-6f) * scale; j = d; }
    else         { base = (float)(pp + 1) / (9.0f + 1e-6f) * scale;  j = d - 128; }
    float freq = powf(10000.0f, (float)(2 * (j / 2)) / 128.0f);
    float v = ((j & 1) == 0) ? sinf(base / freq) : cosf(base / freq);
    _Float16 hv = (_Float16)v;
    for (int b = 0; b < BB; ++b)
        for (int l = 0; l < 3; ++l)
            h[(size_t)(b * NO + l * KK + k) * HCOLS + PD + pp * DD + d] = hv;
}

// ===================================================================
extern "C" void kernel_launch(void* const* d_in, const int* in_sizes, int n_in,
                              void* d_out, int out_size, void* d_ws, size_t ws_size,
                              hipStream_t stream) {
    const float* x[3]   = {(const float*)d_in[0], (const float*)d_in[1], (const float*)d_in[2]};
    const float* obj_w  = (const float*)d_in[3];
    const float* obj_b  = (const float*)d_in[4];
    const float* bn_g   = (const float*)d_in[5];
    const float* bn_b   = (const float*)d_in[6];
    const float* bn_m   = (const float*)d_in[7];
    const float* bn_v   = (const float*)d_in[8];
    const float* offc_w = (const float*)d_in[9];
    const float* offc_b = (const float*)d_in[10];
    const float* offo_w = (const float*)d_in[11];
    const float* offo_b = (const float*)d_in[12];
    const float* fc1_w  = (const float*)d_in[13];
    const float* fc2_w  = (const float*)d_in[14];
    const float* cls_w  = (const float*)d_in[15];
    const float* cls_b  = (const float*)d_in[16];
    const float* bb1_w  = (const float*)d_in[17];
    const float* bb1_b  = (const float*)d_in[18];
    const float* bb2_w  = (const float*)d_in[19];
    const float* bb2_b  = (const float*)d_in[20];
    const float* bb3_w  = (const float*)d_in[21];
    const float* bb3_b  = (const float*)d_in[22];

    float* cls_out = (float*)d_out;                    // (1,B,900,80) = 576000
    float* bb_out  = cls_out + 576000;                 // (1,B,900,4)  = 28800
    float* cent    = bb_out + 28800;                   // centers: 131072+32768+8192
    const int Hs[3] = {128, 64, 32};
    const int centOff[3] = {0, BB * 128 * 128, BB * 128 * 128 + BB * 64 * 64};

    // workspace allocator (256B aligned)
    char* ws = (char*)d_ws;
    size_t off = 0;
    auto A = [&](size_t bytes) -> void* {
        void* p = ws + off;
        off = (off + bytes + 255) & ~(size_t)255;
        return p;
    };
    _Float16* offcW16 = (_Float16*)A((size_t)DD * PD * 2);
    _Float16* fc1T    = (_Float16*)A((size_t)DD * HCOLS * 2);
    _Float16* fc2T    = (_Float16*)A((size_t)DD * DD * 2);
    _Float16* clsT    = (_Float16*)A((size_t)128 * DD * 2);   // 80 -> pad 128 rows
    _Float16* bb1T    = (_Float16*)A((size_t)DD * DD * 2);
    _Float16* bb2T    = (_Float16*)A((size_t)DD * DD * 2);
    _Float16* bb3T    = (_Float16*)A((size_t)64 * DD * 2);    // 4 -> pad 64 rows
    int*      idxs    = (int*)A((size_t)3 * BB * KK * 4);
    float*    xys     = (float*)A((size_t)3 * BB * KK * 2 * 4);
    _Float16* patches = (_Float16*)A((size_t)BB * KK * PD * 2);
    _Float16* offco   = (_Float16*)A((size_t)BB * KK * DD * 2);
    float*    grids   = (float*)A((size_t)3 * BB * KK * PP * 2 * 4);
    _Float16* h       = (_Float16*)A((size_t)BB * NO * HCOLS * 2);
    _Float16* h1      = (_Float16*)A((size_t)BB * NO * DD * 2);
    _Float16* h2      = (_Float16*)A((size_t)BB * NO * DD * 2);
    _Float16* t1      = (_Float16*)A((size_t)BB * NO * DD * 2);
    _Float16* t2      = (_Float16*)A((size_t)BB * NO * DD * 2);

    auto blk = [](int n) { return (n + 255) / 256; };
    auto gemm = [&](const _Float16* Am, const _Float16* Bt, const float* bias,
                    _Float16* oh, float* of, int M, int N16, int Kd,
                    int lda, int ldb, int ldo, int Nreal, int act) {
        int tiles = (M / 16) * (N16 / 64);
        k_wmma_gemm<<<(tiles + 7) / 8, 256, 0, stream>>>(
            Am, Bt, bias, oh, of, M, N16, Kd, lda, ldb, ldo, Nreal, act);
    };

    // --- weight conversion (Bt padded to multiples of 64 rows) ---
    k_copy16<<<blk(DD * PD), 256, 0, stream>>>(offc_w, offcW16, DD * PD);
    k_transpose16<<<blk(DD * HCOLS), 256, 0, stream>>>(fc1_w, fc1T, HCOLS, DD, DD);
    k_transpose16<<<blk(DD * DD), 256, 0, stream>>>(fc2_w, fc2T, DD, DD, DD);
    k_transpose16<<<blk(128 * DD), 256, 0, stream>>>(cls_w, clsT, DD, NCC, 128);
    k_transpose16<<<blk(DD * DD), 256, 0, stream>>>(bb1_w, bb1T, DD, DD, DD);
    k_transpose16<<<blk(DD * DD), 256, 0, stream>>>(bb2_w, bb2T, DD, DD, DD);
    k_transpose16<<<blk(64 * DD), 256, 0, stream>>>(bb3_w, bb3T, DD, 4, 64);

    // --- objectness heads + top-k ---
    for (int l = 0; l < 3; ++l) {
        int HW = Hs[l] * Hs[l];
        k_obj_head<<<blk(BB * HW), 256, 0, stream>>>(
            x[l], obj_w, obj_b, bn_g, bn_b, bn_m, bn_v, l, Hs[l], Hs[l], cent + centOff[l]);
        k_topk<<<BB, 1024, 0, stream>>>(cent + centOff[l], HW, Hs[l], Hs[l],
                                        idxs + l * BB * KK, xys + l * BB * KK * 2);
    }

    // --- per-level: im2col gather, offset conv GEMM, grid, sample ---
    for (int l = 0; l < 3; ++l) {
        k_gather_patches<<<blk(BB * KK * PD), 256, 0, stream>>>(
            x[l], idxs + l * BB * KK, Hs[l], Hs[l], patches);
        gemm(patches, offcW16, offc_b, offco, nullptr,
             BB * KK, DD, PD, PD, PD, DD, DD, /*relu*/1);
        k_offsets_to_grid<<<blk(BB * KK * PP), 256, 0, stream>>>(
            offco, offo_w, offo_b, xys + l * BB * KK * 2, grids + l * BB * KK * PP * 2);
        k_sample_to_h<<<blk(BB * KK * PP * DD), 256, 0, stream>>>(
            x[l], grids + l * BB * KK * PP * 2, l, Hs[l], Hs[l], h);
    }
    k_sine_to_h<<<blk(KK * PP * DD), 256, 0, stream>>>(h);

    // --- MLP head GEMMs ---
    gemm(h,  fc1T, nullptr, h1, nullptr, BB * NO, DD, HCOLS, HCOLS, HCOLS, DD, DD, 1);
    gemm(h1, fc2T, nullptr, h2, nullptr, BB * NO, DD, DD, DD, DD, DD, DD, 1);
    gemm(h2, clsT, cls_b, nullptr, cls_out, BB * NO, 128, DD, DD, DD, NCC, NCC, 0);
    gemm(h2, bb1T, bb1_b, t1, nullptr, BB * NO, DD, DD, DD, DD, DD, DD, 1);
    gemm(t1, bb2T, bb2_b, t2, nullptr, BB * NO, DD, DD, DD, DD, DD, DD, 1);
    gemm(t2, bb3T, bb3_b, nullptr, bb_out, BB * NO, 64, DD, DD, DD, 4, 4, 2);
}
